// SVDQW4A4Linear_33990371180875
// MI455X (gfx1250) — 8-bit matrix operand order repaired and checked by reading, compiled, not run on hardware
//
#include <hip/hip_runtime.h>
#include <hip/hip_bf16.h>

typedef int   v8i __attribute__((ext_vector_type(8)));
typedef float v8f __attribute__((ext_vector_type(8)));
typedef float v2f __attribute__((ext_vector_type(2)));

#define GROUPSZ 64
#define QMAXF 7.0f

__global__ void inv_smooth_kernel(const float* __restrict__ smooth, float* __restrict__ inv_s, int n) {
    int i = blockIdx.x * blockDim.x + threadIdx.x;
    if (i < n) inv_s[i] = 1.0f / smooth[i];
}

__global__ void w4_repack_kernel(const int* __restrict__ qw, signed char* __restrict__ qw8,
                                 long long n) {
    long long i = (long long)blockIdx.x * blockDim.x + threadIdx.x;
    long long stride = (long long)gridDim.x * blockDim.x;
    for (; i < n; i += stride) {
        int K = (int)(i & (GROUPSZ - 1));
        long long base = i - K;
        int dest = ((K >> 3) & 1) * 32 + (K >> 4) * 8 + (K & 7);
        qw8[base + dest] = (signed char)(qw[i] - 8);
    }
}

__global__ __launch_bounds__(256)
void quant_lora_kernel(const float* __restrict__ x,
                       const float* __restrict__ inv_s,
                       const float* __restrict__ pd,
                       signed char* __restrict__ qact,
                       float* __restrict__ ascale_t,
                       float* __restrict__ lora_act,
                       int M, int C, int G, int R) {
    __shared__ float lred[8][32];
    const int m    = blockIdx.x;
    const int wave = threadIdx.x >> 5;
    const int lane = threadIdx.x & 31;

    float lacc[32];
#pragma unroll
    for (int r = 0; r < 32; ++r) lacc[r] = 0.f;

    for (int g = wave; g < G; g += 8) {
        const int c = g * GROUPSZ + lane * 2;
        const float2 xv = *(const float2*)(x + (size_t)m * C + c);
        const float2 sv = *(const float2*)(inv_s + c);
        const float v0 = xv.x * sv.x;
        const float v1 = xv.y * sv.y;

        float amax = fmaxf(fabsf(v0), fabsf(v1));
#pragma unroll
        for (int off = 16; off >= 1; off >>= 1)
            amax = fmaxf(amax, __shfl_xor(amax, off, 32));
        const float ascale = amax / QMAXF;
        if (lane == 0) ascale_t[(size_t)g * M + m] = ascale;

        const float inv = 1.0f / fmaxf(ascale, 1e-8f);
        int q0 = (int)rintf(v0 * inv); q0 = min(7, max(-8, q0));
        int q1 = (int)rintf(v1 * inv); q1 = min(7, max(-8, q1));
        const int K   = lane * 2;
        const int blk = K >> 3;
        const int dst = (blk & 1) * 32 + (blk >> 1) * 8 + (K & 7);
        *(unsigned short*)(qact + (size_t)m * C + g * GROUPSZ + dst) =
            (unsigned short)((q0 & 0xFF) | ((q1 & 0xFF) << 8));

        const float* pd0 = pd + (size_t)c * R;
#pragma unroll
        for (int r = 0; r < 32; ++r) lacc[r] += v0 * pd0[r] + v1 * pd0[R + r];
    }

#pragma unroll
    for (int b = 16; b >= 1; b >>= 1) {
        const bool hib = (lane & b) != 0;
#pragma unroll
        for (int i = 0; i < b; ++i) {
            float send  = hib ? lacc[i] : lacc[i + b];
            float other = __shfl_xor(send, b, 32);
            lacc[i] = (hib ? lacc[i + b] : lacc[i]) + other;
        }
    }
    lred[wave][lane] = lacc[0];
    __syncthreads();
    if (threadIdx.x < 32) {
        float s = 0.f;
#pragma unroll
        for (int w = 0; w < 8; ++w) s += lred[w][threadIdx.x];
        lora_act[(size_t)m * R + threadIdx.x] = s;
    }
}

__global__ __launch_bounds__(256)
void w4a4_gemm_kernel(const signed char* __restrict__ qact,
                      const signed char* __restrict__ qw8,
                      const float* __restrict__ ascale_t,
                      const float* __restrict__ wscales,
                      const float* __restrict__ lora_act,
                      const float* __restrict__ pu,
                      const float* __restrict__ bias,
                      float* __restrict__ out,
                      int M, int C, int C_out, int G, int R) {
    const int wave = threadIdx.x >> 5;
    const int lane = threadIdx.x & 31;
    const int hi   = lane >> 4;
    const int l16  = lane & 15;

    const int m_base = blockIdx.x * 64  + (wave & 3) * 16;
    const int n_base = blockIdx.y * 128 + (wave >> 2) * 64;

    v8f acc[4];
#pragma unroll
    for (int t = 0; t < 4; ++t)
#pragma unroll
        for (int j = 0; j < 8; ++j) acc[t][j] = 0.f;

    v8i izero;
#pragma unroll
    for (int j = 0; j < 8; ++j) izero[j] = 0;

    const signed char* arow = qact + (size_t)(m_base + l16) * C + hi * 32;
    const signed char* brow = qw8  + (size_t)(n_base + l16) * C + hi * 32;

    for (int g = 0; g < G; ++g) {
        union { v8i v; int4 q[2]; } A;
        const signed char* ap = arow + g * GROUPSZ;
        A.q[0] = *(const int4*)(ap);
        A.q[1] = *(const int4*)(ap + 16);
        __builtin_prefetch(ap + GROUPSZ, 0, 1);

        union { v8f v; float4 f[2]; } AS;
        const float* asp = ascale_t + (size_t)g * M + m_base + hi * 8;
        AS.f[0] = *(const float4*)(asp);
        AS.f[1] = *(const float4*)(asp + 4);

        union { v8i v; int4 q[2]; } Bm[4];
        float ws[4];
#pragma unroll
        for (int t = 0; t < 4; ++t) {
            const signed char* bp = brow + (size_t)t * 16 * C + g * GROUPSZ;
            Bm[t].q[0] = *(const int4*)(bp);
            Bm[t].q[1] = *(const int4*)(bp + 16);
            ws[t] = wscales[(size_t)g * C_out + n_base + t * 16 + l16];
        }
        __builtin_prefetch(brow + g * GROUPSZ + GROUPSZ, 0, 1);

        v8i ires[4];
#pragma unroll
        for (int t = 0; t < 4; ++t)
            ires[t] = __builtin_amdgcn_wmma_i32_16x16x64_iu8(
                true, A.v, true, Bm[t].v, izero, false, false);

#pragma unroll
        for (int t = 0; t < 4; ++t)
#pragma unroll
            for (int j = 0; j < 8; ++j)
                acc[t][j] += (float)ires[t][j] * (AS.v[j] * ws[t]);
    }

    const int mrow = m_base + l16;
#pragma unroll
    for (int kb = 0; kb < 32; kb += 4) {
        const int k0 = kb + hi * 2;
        v2f la;
        la[0] = lora_act[(size_t)mrow * R + k0];
        la[1] = lora_act[(size_t)mrow * R + k0 + 1];
#pragma unroll
        for (int t = 0; t < 4; ++t) {
            const int col = n_base + t * 16 + l16;
            v2f pb;
            pb[0] = pu[(size_t)col * R + k0];
            pb[1] = pu[(size_t)col * R + k0 + 1];
            acc[t] = __builtin_amdgcn_wmma_f32_16x16x4_f32(
                false, la, false, pb, (short)0, acc[t], false, false);
        }
    }

#pragma unroll
    for (int t = 0; t < 4; ++t) {
        const int col = n_base + t * 16 + l16;
        const float bv = bias[col];
#pragma unroll
        for (int j = 0; j < 8; ++j) {
            const int row = m_base + j + 8 * hi;
            out[(size_t)row * C_out + col] = acc[t][j] + bv;
        }
    }
}

extern "C" void kernel_launch(void* const* d_in, const int* in_sizes, int n_in,
                              void* d_out, int out_size, void* d_ws, size_t ws_size,
                              hipStream_t stream) {
    const float* x       = (const float*)d_in[0];
    const int*   qweight = (const int*)  d_in[1];
    const float* wscales = (const float*)d_in[2];
    const float* smooth  = (const float*)d_in[3];
    const float* pd      = (const float*)d_in[4];
    const float* pu      = (const float*)d_in[5];
    const float* bias    = (const float*)d_in[6];
    float* out           = (float*)d_out;

    const int C     = in_sizes[3];
    const int M     = in_sizes[0] / C;
    const int C_out = in_sizes[6];
    const int R     = in_sizes[4] / C;
    const int G     = C / GROUPSZ;

    char* ws = (char*)d_ws;
    size_t off = 0;
    auto take = [&](size_t bytes) -> void* {
        void* p = ws + off;
        off += (bytes + 255) & ~(size_t)255;
        return p;
    };
    signed char* qact     = (signed char*)take((size_t)M * C);
    signed char* qw8      = (signed char*)take((size_t)C_out * C);
    float*       ascale_t = (float*)take((size_t)G * M * sizeof(float));
    float*       lora     = (float*)take((size_t)M * R * sizeof(float));
    float*       inv_s    = (float*)take((size_t)C * sizeof(float));
    (void)ws_size; (void)n_in; (void)out_size;

    inv_smooth_kernel<<<(C + 255) / 256, 256, 0, stream>>>(smooth, inv_s, C);
    w4_repack_kernel<<<1024, 256, 0, stream>>>(qweight, qw8, (long long)C_out * C);
    quant_lora_kernel<<<M, 256, 0, stream>>>(x, inv_s, pd, qact, ascale_t, lora, M, C, G, R);
    dim3 grid(M / 64, C_out / 128);
    w4a4_gemm_kernel<<<grid, 256, 0, stream>>>(qact, qw8, ascale_t, wscales, lora, pu, bias, out,
                                               M, C, C_out, G, R);
}
